// Bahdanau_61684320305411
// MI455X (gfx1250) — matrix operand order repaired and checked by reading, not compiled, not run on hardware
//
#include <hip/hip_runtime.h>
#include <hip/hip_bf16.h>

// ---- CDNA5 WMMA types ------------------------------------------------------
typedef __attribute__((ext_vector_type(16))) __bf16       v16bf;
typedef __attribute__((ext_vector_type(8)))  float        v8f;
typedef __attribute__((ext_vector_type(4)))  unsigned int v4u;

union FragAB {           // 32 bytes: one bf16 A/B fragment per lane
  v4u   u[2];
  v16bf v;
};

#define B_  16
#define L_  64
#define D_  512
#define O_  1024
#define R_  (2 * B_ * L_)   // 2048 rows: [q0 rows | q1 rows]
#define NKSTEP (D_ / 32)    // 16 K-steps of 32

#if __has_builtin(__builtin_amdgcn_tanhf)
#define TANH_F32(x) __builtin_amdgcn_tanhf(x)
#else
#define TANH_F32(x) tanhf(x)
#endif

// f32 -> bf16 round-to-nearest-even
__device__ __forceinline__ unsigned short f2bf(float f) {
  unsigned int u = __float_as_uint(f);
  u += 0x7FFFu + ((u >> 16) & 1u);
  return (unsigned short)(u >> 16);
}

// 16B global -> LDS async copy (ASYNCcnt path, no VGPR round trip)
__device__ __forceinline__ void async_cp16(unsigned ldsOff, const void* g) {
  asm volatile("global_load_async_to_lds_b128 %0, %1, off"
               :: "v"(ldsOff), "v"(g) : "memory");
}

// ---- conversions -----------------------------------------------------------
__global__ void cvt_q(const float* __restrict__ q0, const float* __restrict__ q1,
                      unsigned short* __restrict__ qcat) {
  const int n = R_ * D_;
  const int half = B_ * L_ * D_;
  for (int i = blockIdx.x * blockDim.x + threadIdx.x; i < n;
       i += gridDim.x * blockDim.x) {
    float f = (i < half) ? q0[i] : q1[i - half];
    qcat[i] = f2bf(f);
  }
}

// W1 (2D x O, row-major f32) -> W1T (O x 2D, bf16) so B-tile loads are contiguous
__global__ void cvt_w1t(const float* __restrict__ W1, unsigned short* __restrict__ W1T) {
  const int n = O_ * 2 * D_;
  for (int i = blockIdx.x * blockDim.x + threadIdx.x; i < n;
       i += gridDim.x * blockDim.x) {
    int o = i / (2 * D_);
    int d = i - o * (2 * D_);
    W1T[i] = f2bf(W1[d * O_ + o]);
  }
}

// ---- main GEMM: S[2048x1024] = qcat @ W1T^T (+ b1 on s0 half) --------------
// block = 128 threads = 4 waves; output tile 64x64.
// Double-buffered LDS tiles (A: 64x32, B: 64x32 bf16) fed by async copies;
// stage k+1 copies overlap stage k ds_load + wmma.
__global__ void __launch_bounds__(128)
gemm_bf16(const unsigned short* __restrict__ qcat,
          const unsigned short* __restrict__ W1T,
          const float* __restrict__ b1,
          float* __restrict__ S) {
  __shared__ __align__(16) unsigned short ldsA[2][64 * 32];
  __shared__ __align__(16) unsigned short ldsB[2][64 * 32];

  const int lane  = threadIdx.x & 31;
  const int wave  = threadIdx.x >> 5;
  const int laneM = lane & 15;      // M (A) / N (B,D) within tile
  const int laneG = lane >> 4;      // half-wave selects K/M half per ISA layout

  const int rowBlock = blockIdx.x * 64;
  const int rowBase  = rowBlock + wave * 16;
  const int nBase    = blockIdx.y * 64;
  const bool upper   = rowBlock >= (B_ * L_);    // s1 rows -> use W1b columns
  const int  kW0     = upper ? D_ : 0;

  const unsigned ldsAbase = (unsigned)(size_t)&ldsA[0][0];
  const unsigned ldsBbase = (unsigned)(size_t)&ldsB[0][0];

  // Stage one 64x32 A tile + 64x32 B tile into LDS buffer `buf`.
  // 128 threads x 2 chunks x 16B = 4KB per tile; 4 async ops per wave.
  auto stage = [&](int buf, int kbase) {
#pragma unroll
    for (int h = 0; h < 2; ++h) {
      const int c   = (int)threadIdx.x + h * 128;   // chunk 0..255
      const int row = c >> 2;
      const int ko  = (c & 3) * 8;
      const unsigned short* g = qcat + (size_t)(rowBlock + row) * D_ + kbase + ko;
      async_cp16(ldsAbase + (unsigned)(buf * (64 * 32) + row * 32 + ko) * 2u, g);
    }
#pragma unroll
    for (int h = 0; h < 2; ++h) {
      const int c   = (int)threadIdx.x + h * 128;
      const int col = c >> 2;
      const int ko  = (c & 3) * 8;
      const unsigned short* g = W1T + (size_t)(nBase + col) * (2 * D_) + kW0 + kbase + ko;
      async_cp16(ldsBbase + (unsigned)(buf * (64 * 32) + col * 32 + ko) * 2u, g);
    }
  };

  v8f acc[4] = {};

  stage(0, 0);
#pragma unroll
  for (int ks = 0; ks < NKSTEP; ++ks) {
    const int buf = ks & 1;
    if (ks + 1 < NKSTEP) {
      stage(buf ^ 1, (ks + 1) * 32);                 // overlap next-stage copies
      asm volatile("s_wait_asynccnt 0x4" ::: "memory");  // this stage's 4 ops done
    } else {
      asm volatile("s_wait_asynccnt 0x0" ::: "memory");
    }
    __syncthreads();                                  // whole tile visible

    // A fragment: lane=M; elems 0..7 -> k_local = 8g..; 8..15 -> 16+8g..
    const unsigned short* As = &ldsA[buf][0];
    FragAB a;
    a.u[0] = *(const v4u*)(As + (wave * 16 + laneM) * 32 + 8 * laneG);
    a.u[1] = *(const v4u*)(As + (wave * 16 + laneM) * 32 + 16 + 8 * laneG);

    // B fragments: lane=N; elems 0..7 -> k_local = 8g..; 8..15 -> 16+8g..
    const unsigned short* Bs = &ldsB[buf][0];
#pragma unroll
    for (int t = 0; t < 4; ++t) {
      FragAB bfr;
      bfr.u[0] = *(const v4u*)(Bs + (t * 16 + laneM) * 32 + 8 * laneG);
      bfr.u[1] = *(const v4u*)(Bs + (t * 16 + laneM) * 32 + 16 + 8 * laneG);
      acc[t] = __builtin_amdgcn_wmma_f32_16x16x32_bf16(
          /*neg_a=*/false, a.v, /*neg_b=*/false, bfr.v,
          /*c_mod=*/(short)0, acc[t], /*reuse_a=*/false, /*reuse_b=*/false);
    }
    __syncthreads();                                  // reads done before next overwrite
  }

  // D layout: VGPR r -> M = r + 8*laneG ; N = laneM
#pragma unroll
  for (int t = 0; t < 4; ++t) {
    const int ncol = nBase + t * 16 + laneM;
    const float bias = upper ? 0.0f : b1[ncol];
#pragma unroll
    for (int r = 0; r < 8; ++r) {
      const int mrow = rowBase + r + 8 * laneG;
      S[(size_t)mrow * O_ + ncol] = acc[t][r] + bias;
    }
  }
}

// ---- scores: attn[b,i,j] = W2 . tanh(s1[b,i]+s0[b,j]+b1) + b2 - 1e8*mask ---
__global__ void __launch_bounds__(256)
scores_kernel(const float* __restrict__ S, const float* __restrict__ W2,
              const float* __restrict__ b2, const int* __restrict__ mask0,
              const int* __restrict__ mask1, float* __restrict__ attn) {
  __shared__ __align__(16) float s1row[O_];
  __shared__ __align__(16) float w2s[O_];
  const int b = blockIdx.x >> 6;
  const int i = blockIdx.x & 63;

  // Stage s1 row + W2 into LDS via async copies: 256 thr x 16B = 4KB each.
  {
    const float* g1 = S + (size_t)(B_ * L_ + b * L_ + i) * O_ + threadIdx.x * 4;
    const float* g2 = W2 + threadIdx.x * 4;
    async_cp16((unsigned)(size_t)&s1row[0] + threadIdx.x * 16u, g1);
    async_cp16((unsigned)(size_t)&w2s[0] + threadIdx.x * 16u, g2);
    asm volatile("s_wait_asynccnt 0x0" ::: "memory");
  }
  __syncthreads();

  const int wave = threadIdx.x >> 5;
  const int lane = threadIdx.x & 31;
  const float bias = b2[0];
  const int m0 = mask0[b * L_ + i];

  for (int jt = 0; jt < 8; ++jt) {
    const int j = (wave << 3) + jt;
    const float* s0row = S + (size_t)(b * L_ + j) * O_;
    float part = 0.0f;
    for (int o = lane; o < O_; o += 32)
      part += w2s[o] * TANH_F32(s1row[o] + s0row[o]);
#pragma unroll
    for (int off = 16; off > 0; off >>= 1)
      part += __shfl_xor(part, off, 32);
    if (lane == 0) {
      float sc = part + bias;
      if (m0 * mask1[b * L_ + j]) sc -= 1.0e8f;
      attn[(size_t)b * L_ * L_ + i * L_ + j] = sc;
    }
  }
}

// ---- out0[b,x,:] = softmax_y(attn[b,x,:]) @ q1[b,:,:] ----------------------
__global__ void __launch_bounds__(256)
out0_kernel(const float* __restrict__ attn, const float* __restrict__ q1,
            float* __restrict__ out0) {
  __shared__ float raw[L_];
  __shared__ float prob[L_];
  __shared__ float red[2];
  const int b = blockIdx.x >> 6;
  const int x = blockIdx.x & 63;
  if (threadIdx.x < L_)
    raw[threadIdx.x] = attn[((size_t)b * L_ + x) * L_ + threadIdx.x];
  __syncthreads();
  if (threadIdx.x == 0) {
    float mx = raw[0];
    for (int y = 1; y < L_; ++y) mx = fmaxf(mx, raw[y]);
    float s = 0.0f;
    for (int y = 0; y < L_; ++y) s += __expf(raw[y] - mx);
    red[0] = mx; red[1] = 1.0f / s;
  }
  __syncthreads();
  if (threadIdx.x < L_)
    prob[threadIdx.x] = __expf(raw[threadIdx.x] - red[0]) * red[1];
  __syncthreads();
  for (int d = threadIdx.x; d < D_; d += blockDim.x) {
    float acc = 0.0f;
    for (int y = 0; y < L_; ++y)
      acc += prob[y] * q1[((size_t)b * L_ + y) * D_ + d];
    out0[((size_t)b * L_ + x) * D_ + d] = acc;
  }
}

// ---- out1[b,y,:] = softmax_x(attn[b,:,y]) @ q0[b,:,:] ----------------------
__global__ void __launch_bounds__(256)
out1_kernel(const float* __restrict__ attn, const float* __restrict__ q0,
            float* __restrict__ out1) {
  __shared__ float raw[L_];
  __shared__ float prob[L_];
  __shared__ float red[2];
  const int b = blockIdx.x >> 6;
  const int y = blockIdx.x & 63;
  if (threadIdx.x < L_)
    raw[threadIdx.x] = attn[((size_t)b * L_ + threadIdx.x) * L_ + y];
  __syncthreads();
  if (threadIdx.x == 0) {
    float mx = raw[0];
    for (int x = 1; x < L_; ++x) mx = fmaxf(mx, raw[x]);
    float s = 0.0f;
    for (int x = 0; x < L_; ++x) s += __expf(raw[x] - mx);
    red[0] = mx; red[1] = 1.0f / s;
  }
  __syncthreads();
  if (threadIdx.x < L_)
    prob[threadIdx.x] = __expf(raw[threadIdx.x] - red[0]) * red[1];
  __syncthreads();
  for (int d = threadIdx.x; d < D_; d += blockDim.x) {
    float acc = 0.0f;
    for (int x = 0; x < L_; ++x)
      acc += prob[x] * q0[((size_t)b * L_ + x) * D_ + d];
    out1[((size_t)b * L_ + y) * D_ + d] = acc;
  }
}

// ---- launcher --------------------------------------------------------------
extern "C" void kernel_launch(void* const* d_in, const int* in_sizes, int n_in,
                              void* d_out, int out_size, void* d_ws, size_t ws_size,
                              hipStream_t stream) {
  const float* q0    = (const float*)d_in[0];
  const float* q1    = (const float*)d_in[1];
  const int*   mask0 = (const int*)d_in[2];
  const int*   mask1 = (const int*)d_in[3];
  const float* W1    = (const float*)d_in[4];
  const float* b1    = (const float*)d_in[5];
  const float* W2    = (const float*)d_in[6];
  const float* b2    = (const float*)d_in[7];

  char* ws = (char*)d_ws;
  unsigned short* qcat = (unsigned short*)ws;                               // 2 MiB
  unsigned short* W1T  = (unsigned short*)(ws + (size_t)R_ * D_ * 2);       // 2 MiB
  float* S    = (float*)(ws + (size_t)R_ * D_ * 2 + (size_t)O_ * 2 * D_ * 2); // 8 MiB
  float* attn = (float*)((char*)S + (size_t)R_ * O_ * 4);                   // 256 KiB

  float* out0 = (float*)d_out;
  float* out1 = out0 + (size_t)B_ * L_ * D_;

  cvt_q  <<<512, 256, 0, stream>>>(q0, q1, qcat);
  cvt_w1t<<<512, 256, 0, stream>>>(W1, W1T);
  gemm_bf16<<<dim3(R_ / 64, O_ / 64), 128, 0, stream>>>(qcat, W1T, b1, S);
  scores_kernel<<<B_ * L_, 256, 0, stream>>>(S, W2, b2, mask0, mask1, attn);
  out0_kernel<<<B_ * L_, 256, 0, stream>>>(attn, q1, out0);
  out1_kernel<<<B_ * L_, 256, 0, stream>>>(attn, q0, out1);
}
